// GIN_17901423690461
// MI455X (gfx1250) — matrix operand order repaired and checked by reading, not compiled, not run on hardware
//
#include <hip/hip_runtime.h>
#include <hip/hip_bf16.h>

#define N    16384
#define DIN  64
#define DOUT 64
#define BK   128          // K-span staged in LDS per iteration
#define AP   136          // Atile pitch (elements): 16B-aligned rows, conflict-scrambled
#define WP   68           // W LDS pitch (floats): avoids bank conflicts in stage-2 dot
#define GP   65           // agg LDS pitch

typedef __attribute__((ext_vector_type(16))) __bf16 v16bf;
typedef __attribute__((ext_vector_type(8)))  float  v8f;
typedef __attribute__((ext_vector_type(4)))  float  f32x4;   // native clang vector (NT-load OK)

__device__ __forceinline__ unsigned short f2bf(float f) {
    __bf16 h = (__bf16)f;
    return __builtin_bit_cast(unsigned short, h);
}

// Stage 0: rewrite X (fp32) into hi/lo bf16 pairs laid out exactly as the
// per-lane WMMA B operand (32x16 bf16): chunk c (K=32c..), tile t (N=16t..),
// lane l holds column n = 16t + (l&15), K = 32c + ((l&16)?16:0) + j, j=0..15.
__global__ void gin_pack_x(const float* __restrict__ X,
                           unsigned short* __restrict__ bhi,
                           unsigned short* __restrict__ blo) {
    int idx = blockIdx.x * blockDim.x + threadIdx.x;   // 0 .. N*DIN-1
    int j = idx & 15;
    int l = (idx >> 4) & 31;
    int t = (idx >> 9) & 3;
    int c = idx >> 11;
    int n = t * 16 + (l & 15);
    int k = c * 32 + ((l & 16) ? 16 : 0) + j;
    float x  = X[(size_t)k * DIN + n];
    __bf16 hi = (__bf16)x;
    float  hf = (float)hi;
    __bf16 lo = (__bf16)(x - hf);
    bhi[idx] = __builtin_bit_cast(unsigned short, hi);
    blo[idx] = __builtin_bit_cast(unsigned short, lo);
}

__device__ __forceinline__ void store4bf(unsigned short* dst,
                                         float a, float b, float c, float d) {
    uint2 p;
    p.x = (unsigned)f2bf(a) | ((unsigned)f2bf(b) << 16);
    p.y = (unsigned)f2bf(c) | ((unsigned)f2bf(d) << 16);
    *(uint2*)dst = p;   // ds_store_b64, 8B aligned
}

// Main: per block, agg[m0:m0+16, 0:64] = A^T X + X  via bf16 hi/lo WMMA,
// then fused out = relu(agg @ W^T + b).
__global__ void __launch_bounds__(128)
gin_main(const float* __restrict__ A, const float* __restrict__ X,
         const float* __restrict__ W, const float* __restrict__ bias,
         const unsigned short* __restrict__ bhi,
         const unsigned short* __restrict__ blo,
         float* __restrict__ out) {
    __shared__ __align__(16) unsigned short Atile[16 * AP];   // A^T tile, bf16
    __shared__ float Wlds[DOUT * WP];
    __shared__ float aggf[16 * GP];

    const int tid  = threadIdx.x;
    const int lane = tid & 31;
    const int wave = tid >> 5;            // n-tile 0..3 (N-offset 16*wave)
    const int m0   = blockIdx.x * 16;

    // Preload W into LDS (covered by the first barrier in the K loop).
    for (int i = tid; i < DOUT * DIN; i += 128) {
        int o = i >> 6, n = i & 63;
        Wlds[o * WP + n] = W[i];
    }

    v8f acc = {0.f, 0.f, 0.f, 0.f, 0.f, 0.f, 0.f, 0.f};

    const int kt = tid >> 2;              // 0..31: k-group of 4 rows
    const int mt = (tid & 3) * 4;         // 0,4,8,12: m-group of 4 cols

    for (int kc = 0; kc < N; kc += BK) {
        __syncthreads();
        // Stage Atile[m][k] = A[kc+k][m0+m] (transpose in LDS), bf16-exact (A is 0/1).
        const float* ab = A + (size_t)(kc + kt * 4) * N + m0 + mt;
        f32x4 v0 = __builtin_nontemporal_load((const f32x4*)(ab));
        f32x4 v1 = __builtin_nontemporal_load((const f32x4*)(ab + N));
        f32x4 v2 = __builtin_nontemporal_load((const f32x4*)(ab + 2 * (size_t)N));
        f32x4 v3 = __builtin_nontemporal_load((const f32x4*)(ab + 3 * (size_t)N));
        store4bf(&Atile[(mt + 0) * AP + kt * 4], v0[0], v1[0], v2[0], v3[0]);
        store4bf(&Atile[(mt + 1) * AP + kt * 4], v0[1], v1[1], v2[1], v3[1]);
        store4bf(&Atile[(mt + 2) * AP + kt * 4], v0[2], v1[2], v2[2], v3[2]);
        store4bf(&Atile[(mt + 3) * AP + kt * 4], v0[3], v1[3], v2[3], v3[3]);
        __syncthreads();

        #pragma unroll
        for (int c2 = 0; c2 < 4; ++c2) {
            // A fragment (16x32 bf16, contiguous K to match packed B): lane<16 -> K {0..15},
            // lane>=16 -> K {16..31}; two ds_load_b128 per lane.
            union { uint4 u[2]; v16bf v; } af;
            const uint4* ap = (const uint4*)&Atile[(lane & 15) * AP + c2 * 32 +
                                                   ((lane & 16) ? 16 : 0)];
            af.u[0] = ap[0];
            af.u[1] = ap[1];          // +8 elements = +16B

            int c = (kc >> 5) + c2;
            size_t boff = ((size_t)(c * 4 + wave) * 32 + lane) * 16;
            v16bf bh = *(const v16bf*)(bhi + boff);   // contiguous 32B per lane
            v16bf bl = *(const v16bf*)(blo + boff);

            acc = __builtin_amdgcn_wmma_f32_16x16x32_bf16(
                      false, af.v, false, bh, (short)0, acc, false, false);
            acc = __builtin_amdgcn_wmma_f32_16x16x32_bf16(
                      false, af.v, false, bl, (short)0, acc, false, false);
        }
    }

    // agg = acc + self X; C/D layout: VGPR r -> M = r + (lane>=16 ? 8:0), N = lane&15.
    const int nl  = lane & 15;
    const int off = (lane & 16) ? 8 : 0;
    #pragma unroll
    for (int r = 0; r < 8; ++r) {
        int m = r + off;
        float self = X[(size_t)(m0 + m) * DIN + wave * 16 + nl];
        aggf[m * GP + wave * 16 + nl] = acc[r] + self;
    }
    __syncthreads();

    // Stage 2: out[m0+m, o] = relu(b[o] + sum_n aggf[m][n] * W[o][n]).
    const int m  = tid >> 3;     // 0..15
    const int og = tid & 7;      // o = og + 8*jj
    float s[8] = {0.f, 0.f, 0.f, 0.f, 0.f, 0.f, 0.f, 0.f};
    for (int n = 0; n < DIN; ++n) {
        float a = aggf[m * GP + n];
        #pragma unroll
        for (int jj = 0; jj < 8; ++jj)
            s[jj] += a * Wlds[(og + 8 * jj) * WP + n];
    }
    #pragma unroll
    for (int jj = 0; jj < 8; ++jj) {
        float h = s[jj] + bias[og + 8 * jj];
        out[(size_t)(m0 + m) * DOUT + og + 8 * jj] = h > 0.f ? h : 0.f;
    }
}

extern "C" void kernel_launch(void* const* d_in, const int* in_sizes, int n_in,
                              void* d_out, int out_size, void* d_ws, size_t ws_size,
                              hipStream_t stream) {
    const float* A    = (const float*)d_in[0];
    const float* X    = (const float*)d_in[1];
    const float* W    = (const float*)d_in[2];
    const float* bias = (const float*)d_in[3];

    unsigned short* bhi = (unsigned short*)d_ws;              // 2 MB
    unsigned short* blo = bhi + (size_t)N * DIN;              // 2 MB

    gin_pack_x<<<(N * DIN) / 256, 256, 0, stream>>>(X, bhi, blo);
    gin_main<<<N / 16, 128, 0, stream>>>(A, X, W, bias, bhi, blo, (float*)d_out);
}
